// DCNv3_Native_44117904064672
// MI455X (gfx1250) — matrix operand order repaired and checked by reading, not compiled, not run on hardware
//
#include <hip/hip_runtime.h>
#include <hip/hip_bf16.h>
#include <cmath>

typedef __attribute__((ext_vector_type(16))) _Float16 v16h;
typedef __attribute__((ext_vector_type(8)))  _Float16 v8h;
typedef __attribute__((ext_vector_type(4)))  _Float16 v4h;
typedef __attribute__((ext_vector_type(8)))  float    v8f;

#define C1i   128
#define C2i   128
#define K2i   9
#define Hi    64
#define Wi    64
#define HWi   4096            // H*W
#define KDIM  1152            // C1*K2
#define APITCH 136            // padded LDS row pitch (f16): 272B = 68 banks -> conflict-minimal

// gfx1250 async global->LDS copy (16B), tracked by ASYNCcnt (CDNA5 ISA, GV mode)
__device__ __forceinline__ void async_copy16(unsigned lds_off, const void* gptr) {
  asm volatile("global_load_async_to_lds_b128 %0, %1, off"
               :: "v"(lds_off), "v"(gptr) : "memory");
}
__device__ __forceinline__ void wait_async0() {
  asm volatile("s_wait_asynccnt 0" ::: "memory");
}
__device__ __forceinline__ unsigned lds_addr(const void* p) {
  return (unsigned)(size_t)p;   // low 32 bits of generic LDS pointer = DS byte offset
}

// ---------------------------------------------------------------------------
// Prep 0: x NCHW fp32 -> xT NHWC f16 (coalesced channel gathers for sampling)
// ---------------------------------------------------------------------------
__global__ __launch_bounds__(256) void k_transpose(const float* __restrict__ x,
                                                   _Float16* __restrict__ xT) {
  int tid = blockIdx.x * 256 + threadIdx.x;      // over 16*128*64*64 = 8388608
  int w = tid & 63;
  int h = (tid >> 6) & 63;
  int c = (tid >> 12) & 127;
  int b = tid >> 19;
  xT[(((size_t)b * HWi + (h << 6) + w) * C1i) + c] = (_Float16)x[tid];
}

// ---------------------------------------------------------------------------
// Prep 1: pack w_off(18) + w_mask(9) -> wB1[32][1152] f16, K' = tap*128 + c.
// ---------------------------------------------------------------------------
__global__ __launch_bounds__(256) void k_prep_w1(const float* __restrict__ w_off,
                                                 const float* __restrict__ b_off,
                                                 const float* __restrict__ w_mask,
                                                 const float* __restrict__ b_mask,
                                                 _Float16* __restrict__ wB1,
                                                 float* __restrict__ bias1) {
  int tid = blockIdx.x * 256 + threadIdx.x;      // over 32*1152 = 36864
  int kp = tid % KDIM;
  int n  = tid / KDIM;
  int t  = kp >> 7;
  int c  = kp & 127;
  float v = 0.0f;
  if (n < 18)      v = w_off [((n        * C1i + c) * K2i) + t];
  else if (n < 27) v = w_mask[(((n - 18) * C1i + c) * K2i) + t];
  wB1[tid] = (_Float16)v;
  if (blockIdx.x == 0 && threadIdx.x < 32) {
    int nn = threadIdx.x;
    float bv = 0.0f;
    if (nn < 18)      bv = b_off[nn];
    else if (nn < 27) bv = b_mask[nn - 18];
    bias1[nn] = bv;
  }
}

// ---------------------------------------------------------------------------
// Prep 2: w_conv (128,128,3,3) -> wB2[128][1152] f16, K' = tap*128 + c
// ---------------------------------------------------------------------------
__global__ __launch_bounds__(256) void k_prep_w2(const float* __restrict__ w_conv,
                                                 _Float16* __restrict__ wB2) {
  int tid = blockIdx.x * 256 + threadIdx.x;      // over 128*1152 = 147456
  int kp = tid % KDIM;
  int o  = tid / KDIM;
  int t  = kp >> 7;
  int c  = kp & 127;
  wB2[tid] = (_Float16)w_conv[((o * C1i + c) * K2i) + t];
}

// ---------------------------------------------------------------------------
// Phase 1: offset/mask conv as implicit GEMM (M=65536, N=32, K=1152).
// Tap-major staging: per tap, async-DMA the full 128-channel im2col slab
// (A: 128px x 128ch) and B (32 x 128) into LDS, then 4 K-chunks of WMMA.
// ---------------------------------------------------------------------------
__global__ __launch_bounds__(256) void k_offmask(const _Float16* __restrict__ xT,
                                                 const _Float16* __restrict__ wB1,
                                                 const float* __restrict__ bias1,
                                                 float* __restrict__ offmask) {
  __shared__ __align__(16) _Float16 As[128][APITCH];
  __shared__ __align__(16) _Float16 Bs[32][APITCH];
  const int tid    = threadIdx.x;
  const int lane   = tid & 31;
  const int wave   = tid >> 5;
  const int laneHi = lane >> 4;
  const int lane15 = lane & 15;
  const int m0      = blockIdx.x * 128;          // 2 image rows
  const int b       = m0 >> 12;
  const int pixbase = m0 & 4095;

  v8f acc0 = {};
  v8f acc1 = {};

  for (int t = 0; t < 9; ++t) {
    const int ty = t / 3, tx = t % 3;
    __syncthreads();
    // ---- A slab: im2col copy (zero-padded) via async DMA; 2 threads / pixel
    {
      int pix   = tid >> 1;
      int halfB = tid & 1;                       // 64 channels each
      int hp = (pixbase + pix) >> 6;
      int wp = (pixbase + pix) & 63;
      int y  = hp - 1 + ty;
      int xc = wp - 1 + tx;
      _Float16* dst = &As[pix][halfB * 64];
      if ((unsigned)y < (unsigned)Hi && (unsigned)xc < (unsigned)Wi) {
        const _Float16* src = xT + (((size_t)b * HWi + (y << 6) + xc) * C1i) + halfB * 64;
        unsigned l = lds_addr(dst);
#pragma unroll
        for (int i = 0; i < 8; ++i) async_copy16(l + i * 16, src + i * 8);
      } else {
        v8h z = {};
#pragma unroll
        for (int i = 0; i < 8; ++i) *(v8h*)(dst + i * 8) = z;
      }
    }
    // ---- B slab: 32 x 128 f16 via async DMA; 8 threads / row
    {
      int n = tid >> 3;
      int q = tid & 7;                           // 16 f16 each
      unsigned l = lds_addr(&Bs[n][q * 16]);
      const _Float16* src = wB1 + (size_t)n * KDIM + t * 128 + q * 16;
      async_copy16(l,      src);
      async_copy16(l + 16, src + 8);
    }
    wait_async0();
    __syncthreads();
    // ---- 4 K-chunks of 32
#pragma unroll
    for (int kb = 0; kb < 4; ++kb) {
      int arow = wave * 16 + lane15;
      int koff = kb * 32 + laneHi * 16;
      v8h alo = *(const v8h*)&As[arow][koff];
      v8h ahi = *(const v8h*)&As[arow][koff + 8];
      v16h a;
#pragma unroll
      for (int i = 0; i < 8; ++i) { a[i] = alo[i]; a[8 + i] = ahi[i]; }
      {
        v8h blo = *(const v8h*)&Bs[lane15][kb * 32 + laneHi * 16];
        v8h bhi = *(const v8h*)&Bs[lane15][kb * 32 + laneHi * 16 + 8];
        v16h bb;
#pragma unroll
        for (int i = 0; i < 8; ++i) { bb[i] = blo[i]; bb[8 + i] = bhi[i]; }
        acc0 = __builtin_amdgcn_wmma_f32_16x16x32_f16(false, a, false, bb, (short)0, acc0, false, false);
      }
      {
        v8h blo = *(const v8h*)&Bs[16 + lane15][kb * 32 + laneHi * 16];
        v8h bhi = *(const v8h*)&Bs[16 + lane15][kb * 32 + laneHi * 16 + 8];
        v16h bb;
#pragma unroll
        for (int i = 0; i < 8; ++i) { bb[i] = blo[i]; bb[8 + i] = bhi[i]; }
        acc1 = __builtin_amdgcn_wmma_f32_16x16x32_f16(false, a, false, bb, (short)0, acc1, false, false);
      }
    }
  }
  // ---- epilogue: +bias, sigmoid on mask channels, zero pad channels
#pragma unroll
  for (int r = 0; r < 8; ++r) {
    int mg = m0 + wave * 16 + r + laneHi * 8;
    {
      int n = lane15;                             // n in [0,16): raw offsets
      offmask[(size_t)mg * 32 + n] = acc0[r] + bias1[n];
    }
    {
      int n = 16 + lane15;
      float v = acc1[r] + bias1[n];
      float o;
      if (n < 18)      o = v;
      else if (n < 27) o = 1.0f / (1.0f + __expf(-v));
      else             o = 0.0f;
      offmask[(size_t)mg * 32 + n] = o;
    }
  }
}

// ---------------------------------------------------------------------------
// Phase 2: modulated bilinear sampling fused with main GEMM
// (M=65536, N=128, K=1152). Tap-major: per tap each wave samples 8 pixels
// x 128 channels (4 ch/lane, b64 gathers), B async-DMA'd, then 4x4 WMMAs.
// ---------------------------------------------------------------------------
__global__ __launch_bounds__(256) void k_deform_gemm(const _Float16* __restrict__ xT,
                                                     const float* __restrict__ offmask,
                                                     const _Float16* __restrict__ wB2,
                                                     float* __restrict__ out) {
  __shared__ __align__(16) _Float16 As[64][APITCH];
  __shared__ __align__(16) _Float16 Bs[128][APITCH];
  const int tid    = threadIdx.x;
  const int lane   = tid & 31;
  const int wave   = tid >> 5;
  const int laneHi = lane >> 4;
  const int lane15 = lane & 15;
  const int m0      = blockIdx.x * 64;           // one full image row
  const int b       = m0 >> 12;
  const int pixbase = m0 & 4095;
  const int h0      = pixbase >> 6;
  const int msub    = wave & 3;
  const int nbase   = (wave >> 2) * 64;

  v8f acc[4] = {};
  const _Float16* xb = xT + (size_t)b * HWi * C1i;

  for (int t = 0; t < 9; ++t) {
    const int ty = t / 3, tx = t % 3;
    __syncthreads();
    // ---- B slab: 128 x 128 f16 via async DMA; 2 threads / row
    {
      int n    = tid >> 1;
      int half = tid & 1;
      unsigned l = lds_addr(&Bs[n][half * 64]);
      const _Float16* src = wB2 + (size_t)n * KDIM + t * 128 + half * 64;
#pragma unroll
      for (int i = 0; i < 8; ++i) async_copy16(l + i * 16, src + i * 8);
    }
    // ---- A slab: modulated bilinear samples; wave handles 8 pixels, 4 ch/lane.
    // Position/weight math once per (pixel, tap); corner branches are wave-uniform.
    for (int p = 0; p < 8; ++p) {
      int pix = wave * 8 + p;
      int mg  = m0 + pix;
      const float* om = offmask + (size_t)mg * 32;
      float offy = om[2 * t];
      float offx = om[2 * t + 1];
      float mk   = om[18 + t];
      float py = offy + (float)(h0 - 1 + ty);
      float px = offx + (float)(pix - 1 + tx);
      float y0f = floorf(py), x0f = floorf(px);
      float wy1 = py - y0f,  wx1 = px - x0f;
      float wy0 = 1.0f - wy1, wx0 = 1.0f - wx1;
      int y0 = (int)y0f, x0 = (int)x0f;
      int c4 = lane * 4;
      float s0 = 0.0f, s1 = 0.0f, s2 = 0.0f, s3 = 0.0f;
      if ((unsigned)y0 < (unsigned)Hi && (unsigned)x0 < (unsigned)Wi) {
        v4h q = *(const v4h*)&xb[(((y0) << 6) + x0) * C1i + c4];
        float wt = wy0 * wx0;
        s0 += wt * (float)q[0]; s1 += wt * (float)q[1]; s2 += wt * (float)q[2]; s3 += wt * (float)q[3];
      }
      if ((unsigned)y0 < (unsigned)Hi && (unsigned)(x0 + 1) < (unsigned)Wi) {
        v4h q = *(const v4h*)&xb[(((y0) << 6) + x0 + 1) * C1i + c4];
        float wt = wy0 * wx1;
        s0 += wt * (float)q[0]; s1 += wt * (float)q[1]; s2 += wt * (float)q[2]; s3 += wt * (float)q[3];
      }
      if ((unsigned)(y0 + 1) < (unsigned)Hi && (unsigned)x0 < (unsigned)Wi) {
        v4h q = *(const v4h*)&xb[(((y0 + 1) << 6) + x0) * C1i + c4];
        float wt = wy1 * wx0;
        s0 += wt * (float)q[0]; s1 += wt * (float)q[1]; s2 += wt * (float)q[2]; s3 += wt * (float)q[3];
      }
      if ((unsigned)(y0 + 1) < (unsigned)Hi && (unsigned)(x0 + 1) < (unsigned)Wi) {
        v4h q = *(const v4h*)&xb[(((y0 + 1) << 6) + x0 + 1) * C1i + c4];
        float wt = wy1 * wx1;
        s0 += wt * (float)q[0]; s1 += wt * (float)q[1]; s2 += wt * (float)q[2]; s3 += wt * (float)q[3];
      }
      v4h r;
      r[0] = (_Float16)(s0 * mk); r[1] = (_Float16)(s1 * mk);
      r[2] = (_Float16)(s2 * mk); r[3] = (_Float16)(s3 * mk);
      *(v4h*)&As[pix][c4] = r;                   // ds_store_b64
    }
    wait_async0();
    __syncthreads();
    // ---- 4 K-chunks x 4 N-subtiles of WMMA
#pragma unroll
    for (int kb = 0; kb < 4; ++kb) {
      int arow = msub * 16 + lane15;
      int koff = kb * 32 + laneHi * 16;
      v8h alo = *(const v8h*)&As[arow][koff];
      v8h ahi = *(const v8h*)&As[arow][koff + 8];
      v16h a;
#pragma unroll
      for (int i = 0; i < 8; ++i) { a[i] = alo[i]; a[8 + i] = ahi[i]; }
#pragma unroll
      for (int j = 0; j < 4; ++j) {
        int nrow = nbase + j * 16 + lane15;
        v8h blo = *(const v8h*)&Bs[nrow][kb * 32 + laneHi * 16];
        v8h bhi = *(const v8h*)&Bs[nrow][kb * 32 + laneHi * 16 + 8];
        v16h bb;
#pragma unroll
        for (int i = 0; i < 8; ++i) { bb[i] = blo[i]; bb[8 + i] = bhi[i]; }
        acc[j] = __builtin_amdgcn_wmma_f32_16x16x32_f16(false, a, false, bb, (short)0, acc[j], false, false);
      }
    }
  }
  // ---- epilogue: out is NCHW fp32
  const size_t outb = (size_t)b * C2i * HWi;
#pragma unroll
  for (int j = 0; j < 4; ++j) {
    int n = nbase + j * 16 + lane15;
#pragma unroll
    for (int r = 0; r < 8; ++r) {
      int pixrow = pixbase + msub * 16 + r + laneHi * 8;
      out[outb + (size_t)n * HWi + pixrow] = acc[j][r];
    }
  }
}

// ---------------------------------------------------------------------------
// Launch: inputs {x, w_conv, w_off, b_off, w_mask, b_mask} fp32.
// Workspace (bytes, 256-aligned):
//   xT f16      : 16,777,216  @ 0
//   offmask f32 :  8,388,608  @ 16,777,216
//   wB1 f16     :     73,728  @ 25,165,824
//   wB2 f16     :    294,912  @ 25,239,552
//   bias1 f32   :        128  @ 25,534,464          (total ~25.5 MB)
// ---------------------------------------------------------------------------
extern "C" void kernel_launch(void* const* d_in, const int* in_sizes, int n_in,
                              void* d_out, int out_size, void* d_ws, size_t ws_size,
                              hipStream_t stream) {
  (void)in_sizes; (void)n_in; (void)out_size; (void)ws_size;
  const float* x      = (const float*)d_in[0];
  const float* w_conv = (const float*)d_in[1];
  const float* w_off  = (const float*)d_in[2];
  const float* b_off  = (const float*)d_in[3];
  const float* w_mask = (const float*)d_in[4];
  const float* b_mask = (const float*)d_in[5];
  float* out = (float*)d_out;

  char* ws = (char*)d_ws;
  _Float16* xT      = (_Float16*)(ws);
  float*    offmask = (float*)   (ws + 16777216);
  _Float16* wB1     = (_Float16*)(ws + 25165824);
  _Float16* wB2     = (_Float16*)(ws + 25239552);
  float*    bias1   = (float*)   (ws + 25534464);

  hipLaunchKernelGGL(k_transpose,   dim3(32768), dim3(256), 0, stream, x, xT);
  hipLaunchKernelGGL(k_prep_w1,     dim3(144),   dim3(256), 0, stream, w_off, b_off, w_mask, b_mask, wB1, bias1);
  hipLaunchKernelGGL(k_prep_w2,     dim3(576),   dim3(256), 0, stream, w_conv, wB2);
  hipLaunchKernelGGL(k_offmask,     dim3(512),   dim3(256), 0, stream, xT, wB1, bias1, offmask);
  hipLaunchKernelGGL(k_deform_gemm, dim3(1024),  dim3(256), 0, stream, xT, offmask, wB2, out);
}
